// SelectiveAttentionGCNLayer_82678120448018
// MI455X (gfx1250) — matrix operand order repaired and checked by reading, not compiled, not run on hardware
//
#include <hip/hip_runtime.h>
#include <hip/hip_bf16.h>

// Problem constants (match reference)
#define B_ 8
#define S_ 1024
#define D_ 512
#define H_ 8
#define DK_ 64
#define L_ 2

typedef __bf16 bf16;
typedef __attribute__((ext_vector_type(16))) __bf16 bf16x16;
typedef __attribute__((ext_vector_type(8))) float f32x8;

union Frag16 { bf16x16 v; uint4 u[2]; };

// ---- WMMA fragment loaders (CDNA5 wave32 layouts, 16-bit data) ----
// A 16x32 (MxK): lane<16 -> row=lane, K = 0..7 & 16..23 ; lane>=16 -> row=lane-16, K = 8..15 & 24..31
__device__ __forceinline__ bf16x16 load_a16(const bf16* base, int ld, int lane) {
    int row = lane & 15;
    int kc  = (lane >> 4) << 3;              // 0 or 8
    const uint4* p = reinterpret_cast<const uint4*>(base + (size_t)row * ld + kc);
    Frag16 f;
    f.u[0] = p[0];                            // K = kc .. kc+7
    f.u[1] = p[2];                            // K = kc+16 .. kc+23
    return f.v;
}
// B 32x16 (KxN) from a source stored as rows of N (i.e. src[n][k] contiguous in k):
// lane<16 -> col=lane, K = 0..7 & 16..23 ; lane>=16 -> col=lane-16, K = 8..15 & 24..31
__device__ __forceinline__ bf16x16 load_b16(const bf16* base, int ld, int lane) {
    int n  = lane & 15;
    int kc = (lane >> 4) << 3;               // 0 or 8
    const uint4* p = reinterpret_cast<const uint4*>(base + (size_t)n * ld + kc);
    Frag16 f;
    f.u[0] = p[0];
    f.u[1] = p[2];
    return f.v;
}

__device__ __forceinline__ f32x8 wmma_bf16(bf16x16 a, bf16x16 b, f32x8 c) {
    return __builtin_amdgcn_wmma_f32_16x16x32_bf16(false, a, false, b, (short)0, c,
                                                   false, false);
}

// ---------------- small prep kernels ----------------

// wT[n*K + k] = (bf16) w[k*N + n]
__global__ void cvt_transpose_kernel(const float* __restrict__ w, bf16* __restrict__ wT,
                                     int K, int N) {
    int i = blockIdx.x * 256 + threadIdx.x;
    if (i < K * N) {
        int n = i / K, k = i % K;
        wT[i] = (bf16)w[(size_t)k * N + n];
    }
}

__global__ void cvt_f32_bf16_kernel(const float* __restrict__ in, bf16* __restrict__ out, int n) {
    int i = blockIdx.x * 256 + threadIdx.x;
    if (i < n) out[i] = (bf16)in[i];
}

// aspect = mean_s(x * aspect_mask); asp_vec = aspect@asp_w + asp_b; aw[b,h,e] = sum_d asp_vec[d]*wm[h,d,e]
__global__ __launch_bounds__(256) void aspect_kernel(
    const float* __restrict__ x, const float* __restrict__ amask,
    const float* __restrict__ asp_w, const float* __restrict__ asp_b,
    const float* __restrict__ wm, float* __restrict__ aw) {
    __shared__ float s_aspect[D_];
    __shared__ float s_vec[DK_];
    int b = blockIdx.x, tid = threadIdx.x;
    for (int d = tid; d < D_; d += 256) {
        float sum = 0.f;
        for (int s = 0; s < S_; ++s)
            sum += x[((size_t)b * S_ + s) * D_ + d] * amask[(size_t)b * S_ + s];
        s_aspect[d] = sum * (1.0f / (float)S_);
    }
    __syncthreads();
    if (tid < DK_) {
        float sum = asp_b[tid];
        for (int d = 0; d < D_; ++d) sum += s_aspect[d] * asp_w[(size_t)d * DK_ + tid];
        s_vec[tid] = sum;
    }
    __syncthreads();
    for (int idx = tid; idx < H_ * DK_; idx += 256) {
        int h = idx >> 6, e = idx & 63;
        float sum = 0.f;
        for (int d = 0; d < DK_; ++d) sum += s_vec[d] * wm[((size_t)h * DK_ + d) * DK_ + e];
        aw[((size_t)b * H_ + h) * DK_ + e] = sum;
    }
}

// aspsc[b,h,s] = tanh( dot(aw[b,h,:], k[b,s,h*DK:...]) + bm )
__global__ __launch_bounds__(256) void asp_score_kernel(
    const bf16* __restrict__ Kb, const float* __restrict__ aw,
    const float* __restrict__ bm, float* __restrict__ aspsc) {
    int idx = blockIdx.x * 256 + threadIdx.x;        // (b*H + h)*S + s
    int s = idx % S_;
    int h = (idx / S_) % H_;
    int b = idx / (S_ * H_);
    const bf16* kr = Kb + ((size_t)b * S_ + s) * D_ + h * DK_;
    const float* a = aw + ((size_t)b * H_ + h) * DK_;
    float sum = bm[0];
#pragma unroll 8
    for (int d = 0; d < DK_; ++d) sum += a[d] * (float)kr[d];
    aspsc[idx] = tanhf(sum);
}

// ---------------- WMMA GEMM: C[M,N] = A[M,K] @ W[K,N] + bias ----------------
// A bf16 row-major (ld=K), WT bf16 stored as WT[n][k] (ld=K). Optional transposed
// store for v: out[b, n, s] with s = m % Sdim.
template <bool TROUT>
__global__ __launch_bounds__(256) void gemm_bf16_wmma(
    const bf16* __restrict__ A, const bf16* __restrict__ WT,
    const float* __restrict__ bias, bf16* __restrict__ C,
    int M, int N, int K, int Sdim) {
    int w = threadIdx.x >> 5, lane = threadIdx.x & 31;
    int rb = blockIdx.x * 128 + w * 16;   // 8 waves x 16 rows
    int nb = blockIdx.y * 64;
    const f32x8 zero8 = {0.f, 0.f, 0.f, 0.f, 0.f, 0.f, 0.f, 0.f};
    f32x8 acc[4] = {zero8, zero8, zero8, zero8};

    for (int kb = 0; kb < K; kb += 32) {
        bf16x16 a = load_a16(A + (size_t)rb * K + kb, K, lane);
#pragma unroll
        for (int c = 0; c < 4; ++c) {
            bf16x16 bf = load_b16(WT + (size_t)(nb + c * 16) * K + kb, K, lane);
            acc[c] = wmma_bf16(a, bf, acc[c]);
        }
        if (kb + 32 < K)
            __builtin_prefetch(A + (size_t)rb * K + kb + 32, 0, 1);  // global_prefetch_b8
    }
    int n0 = lane & 15, half = lane >> 4;
#pragma unroll
    for (int c = 0; c < 4; ++c) {
        int n = nb + c * 16 + n0;
        float bv = bias[n];
#pragma unroll
        for (int j = 0; j < 8; ++j) {
            int m = rb + j + 8 * half;
            float val = acc[c][j] + bv;
            if (TROUT) {
                int s = m % Sdim, bb = m / Sdim;
                C[((size_t)bb * N + n) * Sdim + s] = (bf16)val;
            } else {
                C[(size_t)m * N + n] = (bf16)val;
            }
        }
    }
}

// ---------------- fused flash attention (per (b,h), 128 q-rows / block) ----------------
// K/V tiles are staged into LDS once per block with async global->LDS loads
// (ASYNCcnt-tracked, double-buffered) instead of 8x-redundant per-wave loads.
__global__ __launch_bounds__(256) void flash_attn_wmma(
    const bf16* __restrict__ Q, const bf16* __restrict__ Kb,
    const bf16* __restrict__ VT, const float* __restrict__ aspsc,
    const unsigned char* __restrict__ srcmask, const float* __restrict__ bias,
    float* __restrict__ Xout, bf16* __restrict__ XBout) {
    __shared__ bf16 ldsK[2][32][64];                 // keys(32) x d(64), double buffered
    __shared__ bf16 ldsV[2][64][32];                 // d(64) x keys(32), double buffered
    __shared__ bf16 ldsP[8][16][32];                 // per-wave P staging (8 KB)
    const int QB = S_ / 128;                         // 8 query tiles
    int bid = blockIdx.x;
    int qblk = bid % QB;
    int h = (bid / QB) % H_;
    int b = bid / (QB * H_);
    int tid = threadIdx.x;
    int w = tid >> 5, lane = tid & 31;
    int s0 = qblk * 128 + w * 16;
    int n0 = lane & 15, half = lane >> 4;

    const bf16* qbase = Q + ((size_t)b * S_ + s0) * D_ + h * DK_;
    bf16x16 qa0 = load_a16(qbase, D_, lane);         // q rows, d 0..31
    bf16x16 qa1 = load_a16(qbase + 32, D_, lane);    // q rows, d 32..63

    const f32x8 zero8 = {0.f, 0.f, 0.f, 0.f, 0.f, 0.f, 0.f, 0.f};
    f32x8 o[4] = {zero8, zero8, zero8, zero8};
    float mrow[8], lrow[8];
#pragma unroll
    for (int j = 0; j < 8; ++j) { mrow[j] = -3.0e38f; lrow[j] = 0.f; }

    const float scale = 0.125f;                      // 1/sqrt(DK)
    const float* aspRow = aspsc + ((size_t)b * H_ + h) * S_;
    const unsigned char* maskRow = srcmask + (size_t)b * S_;
    const bf16* vbase = VT + (size_t)b * DK_ * S_;

    // per-thread async staging: 1x b128 of K tile + 1x b128 of V tile per block step
    int krow = tid >> 3, kseg = (tid & 7) * 8;       // K tile: 32 rows x 64 elems
    int vrow = tid >> 2, vseg = (tid & 3) * 8;       // V tile: 64 rows x 32 elems
    auto issue_async = [&](int kb, int pb) {
        const bf16* kg = Kb + ((size_t)b * S_ + kb + krow) * D_ + h * DK_ + kseg;
        unsigned kl = (unsigned)(uintptr_t)&ldsK[pb][krow][kseg];
        asm volatile("global_load_async_to_lds_b128 %0, %1, off"
                     :: "v"(kl), "v"((unsigned long long)(uintptr_t)kg) : "memory");
        const bf16* vg = vbase + (size_t)vrow * S_ + kb + vseg;
        unsigned vl = (unsigned)(uintptr_t)&ldsV[pb][vrow][vseg];
        asm volatile("global_load_async_to_lds_b128 %0, %1, off"
                     :: "v"(vl), "v"((unsigned long long)(uintptr_t)vg) : "memory");
    };

    issue_async(0, 0);
    for (int kb = 0; kb < S_; kb += 32) {
        int pb = (kb >> 5) & 1;
        if (kb + 32 < S_) {
            issue_async(kb + 32, pb ^ 1);            // prefetch next tile into other buffer
            asm volatile("s_wait_asynccnt 0x2" ::: "memory");  // current tile done (in-order)
        } else {
            asm volatile("s_wait_asynccnt 0x0" ::: "memory");
        }
        __syncthreads();                             // LDS tile visible to all waves

        // scores: two 16x16 tiles (keys kb..kb+15, kb+16..kb+31), K=DK in 2 chunks
        f32x8 sA = zero8, sB = zero8;
        bf16x16 k00 = load_b16(&ldsK[pb][0][0], 64, lane);
        bf16x16 k01 = load_b16(&ldsK[pb][0][0] + 32, 64, lane);
        bf16x16 k10 = load_b16(&ldsK[pb][16][0], 64, lane);
        bf16x16 k11 = load_b16(&ldsK[pb][16][0] + 32, 64, lane);
        sA = wmma_bf16(qa0, k00, sA);
        sA = wmma_bf16(qa1, k01, sA);
        sB = wmma_bf16(qa0, k10, sB);
        sB = wmma_bf16(qa1, k11, sB);

        float asp0 = aspRow[kb + n0];
        float asp1 = aspRow[kb + 16 + n0];
        bool keep0 = (maskRow[kb + n0] == 0);        // keep where ~src_mask
        bool keep1 = (maskRow[kb + 16 + n0] == 0);

#pragma unroll
        for (int j = 0; j < 8; ++j) {
            float sv0 = keep0 ? (sA[j] * scale + asp0) : -1e18f;
            float sv1 = keep1 ? (sB[j] * scale + asp1) : -1e18f;
            // row max across the 16 lanes holding this row
            float t = fmaxf(sv0, sv1);
#pragma unroll
            for (int d = 1; d < 16; d <<= 1) t = fmaxf(t, __shfl_xor(t, d, 32));
            float mnew = fmaxf(mrow[j], t);
            float sc = __expf(mrow[j] - mnew);
            float p0 = __expf(sv0 - mnew);
            float p1 = __expf(sv1 - mnew);
            float rs = p0 + p1;
#pragma unroll
            for (int d = 1; d < 16; d <<= 1) rs += __shfl_xor(rs, d, 32);
            lrow[j] = lrow[j] * sc + rs;
            mrow[j] = mnew;
#pragma unroll
            for (int c = 0; c < 4; ++c) o[c][j] *= sc;
            int row = j + 8 * half;                  // C-layout row for this lane/slot
            ldsP[w][row][n0] = (bf16)p0;
            ldsP[w][row][n0 + 16] = (bf16)p1;
        }

        // o += P(16x32) @ V(32x64); V fragments contiguous thanks to vT tile layout
        bf16x16 pa = load_a16(&ldsP[w][0][0], 32, lane);
#pragma unroll
        for (int c = 0; c < 4; ++c) {
            bf16x16 fv = load_b16(&ldsV[pb][c * 16][0], 32, lane);
            o[c] = wmma_bf16(pa, fv, o[c]);
        }
        __syncthreads();                             // all waves done with tile pb
    }

    // epilogue: normalize, +bias, relu, write fp32 + bf16 copies
#pragma unroll
    for (int j = 0; j < 8; ++j) {
        float inv = 1.0f / lrow[j];
        int s = s0 + j + 8 * half;
        size_t base = ((size_t)b * S_ + s) * D_ + h * DK_;
#pragma unroll
        for (int c = 0; c < 4; ++c) {
            int d = c * 16 + n0;
            float val = fmaxf(o[c][j] * inv + bias[d], 0.0f);
            Xout[base + d] = val;
            XBout[base + d] = (bf16)val;
        }
    }
}

// ---------------- host side ----------------
extern "C" void kernel_launch(void* const* d_in, const int* in_sizes, int n_in,
                              void* d_out, int out_size, void* d_ws, size_t ws_size,
                              hipStream_t stream) {
    const float* x        = (const float*)d_in[0];
    // d_in[1] adj: unused by forward
    const unsigned char* src_mask = (const unsigned char*)d_in[2];   // bool array
    const float* amask    = (const float*)d_in[3];
    const float* wq       = (const float*)d_in[4];
    const float* bq       = (const float*)d_in[5];
    const float* wk       = (const float*)d_in[6];
    const float* bk       = (const float*)d_in[7];
    const float* asp_w    = (const float*)d_in[8];
    const float* asp_b    = (const float*)d_in[9];
    const float* wm       = (const float*)d_in[10];
    const float* bm       = (const float*)d_in[11];
    const float* dense_w  = (const float*)d_in[12];
    const float* dense_b  = (const float*)d_in[13];
    const float* bias     = (const float*)d_in[14];
    float* out            = (float*)d_out;

    // workspace layout (bytes, 256-aligned)
    char* ws = (char*)d_ws;
    size_t off = 0;
    auto take = [&](size_t bytes) { char* p = ws + off; off = (off + bytes + 255) & ~(size_t)255; return p; };
    bf16*  XB   = (bf16*) take((size_t)B_ * S_ * D_ * 2);      // bf16 x (layer input)
    bf16*  QB   = (bf16*) take((size_t)B_ * S_ * D_ * 2);
    bf16*  KB   = (bf16*) take((size_t)B_ * S_ * D_ * 2);
    bf16*  VT   = (bf16*) take((size_t)B_ * DK_ * S_ * 2);     // v transposed [B,DK,S]
    float* X1   = (float*)take((size_t)B_ * S_ * D_ * 4);      // layer-0 fp32 output
    bf16*  WQT  = (bf16*) take((size_t)D_ * D_ * 2);
    bf16*  WKT  = (bf16*) take((size_t)D_ * D_ * 2);
    bf16*  DWT  = (bf16*) take((size_t)DK_ * D_ * 2);
    float* AW   = (float*)take((size_t)B_ * H_ * DK_ * 4);
    float* ASPS = (float*)take((size_t)B_ * H_ * S_ * 4);

    // one-time prep
    cvt_transpose_kernel<<<(D_ * D_ + 255) / 256, 256, 0, stream>>>(wq, WQT, D_, D_);
    cvt_transpose_kernel<<<(D_ * D_ + 255) / 256, 256, 0, stream>>>(wk, WKT, D_, D_);
    cvt_transpose_kernel<<<(D_ * DK_ + 255) / 256, 256, 0, stream>>>(dense_w, DWT, D_, DK_);
    aspect_kernel<<<B_, 256, 0, stream>>>(x, amask, asp_w, asp_b, wm, AW);
    cvt_f32_bf16_kernel<<<(B_ * S_ * D_) / 256, 256, 0, stream>>>(x, XB, B_ * S_ * D_);

    const int M = B_ * S_;
    for (int layer = 0; layer < L_; ++layer) {
        // projections (bf16 WMMA, f32 accumulate)
        gemm_bf16_wmma<false><<<dim3(M / 128, D_ / 64), 256, 0, stream>>>(
            XB, WQT, bq, QB, M, D_, D_, S_);
        gemm_bf16_wmma<false><<<dim3(M / 128, D_ / 64), 256, 0, stream>>>(
            XB, WKT, bk, KB, M, D_, D_, S_);
        gemm_bf16_wmma<true><<<dim3(M / 128, 1), 256, 0, stream>>>(
            XB, DWT, dense_b, VT, M, DK_, D_, S_);
        // aspect scores from projected keys
        asp_score_kernel<<<(B_ * H_ * S_) / 256, 256, 0, stream>>>(KB, AW, bm, ASPS);
        // fused attention; writes next-layer fp32 x and bf16 x
        float* xout = (layer == 0) ? X1 : out;
        flash_attn_wmma<<<B_ * H_ * (S_ / 128), 256, 0, stream>>>(
            QB, KB, VT, ASPS, src_mask, bias + (size_t)layer * DK_, xout, XB);
    }
    (void)in_sizes; (void)n_in; (void)out_size; (void)ws_size; (void)X1;
}
